// yAwareSimCLR_52183852646963
// MI455X (gfx1250) — matrix operand order repaired and checked by reading, not compiled, not run on hardware
//
#include <hip/hip_runtime.h>
#include <hip/hip_bf16.h>

typedef __attribute__((ext_vector_type(16))) __bf16 v16bf;
typedef __attribute__((ext_vector_type(8)))  __bf16 bf16x8;
typedef __attribute__((ext_vector_type(8)))  float  v8f;

#define TEMP_INV 10.0f          // 1 / 0.1
#define GAMMA_C  0.02f          // 1 / (2 * 5^2)
#define INF_C    1.0e8f
#define BN_EPS_C 1e-5f

#define WMMA_BF16(a, b, c) \
  __builtin_amdgcn_wmma_f32_16x16x32_bf16(false, (a), false, (b), (short)0, (c), false, false)

// WGP-scope prefetch (scope field omitted -> 0 = WGP: pull into all cache
// levels on miss). __builtin_prefetch can only reach SE/SYS scope, which
// skips the WGP cache entirely -- useless for L2-resident operands.
__device__ __forceinline__ void prefetch_wgp(const void* p) {
  asm volatile("global_prefetch_b8 %0, off" :: "v"(p));
}

// ---------------------------------------------------------------------------
// WMMA fragment loaders (bf16 16x16x32, contiguous k = 16h + s in both A and B)
// A (16xK tile of row-major X):  m = lane%16 ; K elems 16h..16h+15
// B (Kx16 tile of W^T, W row-major [cols,K]): n = lane%16 ; K elems 16h..16h+15
// ---------------------------------------------------------------------------
__device__ __forceinline__ v16bf load_frag_a(const __bf16* __restrict__ base, int ld,
                                             int row0, int kk, int lane) {
  const int half = lane >> 4, m = lane & 15;
  const __bf16* p = base + (size_t)(row0 + m) * ld + kk + half * 16;
  bf16x8 lo = *(const bf16x8*)(p);
  bf16x8 hi = *(const bf16x8*)(p + 8);
  v16bf f;
#pragma unroll
  for (int e = 0; e < 8; ++e) { f[e] = lo[e]; f[e + 8] = hi[e]; }
  return f;
}

__device__ __forceinline__ v16bf load_frag_b(const __bf16* __restrict__ base, int ld,
                                             int col0, int kk, int lane) {
  const int half = lane >> 4, n = lane & 15;
  return *(const v16bf*)(base + (size_t)(col0 + n) * ld + kk + half * 16);
}

// One K=32 step of the 16x64-per-wave register-blocked GEMM.
__device__ __forceinline__ void gemm_step(const __bf16* __restrict__ A,
                                          const __bf16* __restrict__ B, int K,
                                          int row0, int col0, int kk, int lane,
                                          v8f& c0, v8f& c1, v8f& c2, v8f& c3) {
  v16bf a  = load_frag_a(A, K, row0, kk, lane);
  v16bf b0 = load_frag_b(B, K, col0,      kk, lane);
  v16bf b1 = load_frag_b(B, K, col0 + 16, kk, lane);
  v16bf b2 = load_frag_b(B, K, col0 + 32, kk, lane);
  v16bf b3 = load_frag_b(B, K, col0 + 48, kk, lane);
  c0 = WMMA_BF16(a, b0, c0);
  c1 = WMMA_BF16(a, b1, c1);
  c2 = WMMA_BF16(a, b2, c2);
  c3 = WMMA_BF16(a, b3, c3);
}

// ---------------------------------------------------------------------------
// Elementwise fp32 -> bf16
// ---------------------------------------------------------------------------
__global__ void cvt_bf16(const float* __restrict__ in, __bf16* __restrict__ out, size_t n) {
  size_t i = (size_t)blockIdx.x * blockDim.x + threadIdx.x;
  if (i < n) out[i] = (__bf16)in[i];
}

__global__ void zero_f32(float* __restrict__ p, int n) {
  int i = blockIdx.x * blockDim.x + threadIdx.x;
  if (i < n) p[i] = 0.0f;
}

// ---------------------------------------------------------------------------
// C[M,Ncols] = A[M,K] * B[Ncols,K]^T   (bf16 in, fp32 accumulate/out)
// Register-blocked: each wave computes a 16x64 strip (1 A frag -> 4 WMMAs).
// block = 256 threads (8 waves) -> 128x64 of C per block; all 8 waves share
// the same B panel (hits WGP cache after first wave's miss).
// Manually unrolled x2 over K (K % 64 == 0): 8 WMMAs + 3 prefetches per
// iteration, single basic block (asm prefetch otherwise inhibits unrolling).
// grid = (M/128, Ncols/64)
// ---------------------------------------------------------------------------
__global__ void gemm_xwT(const __bf16* __restrict__ A, const __bf16* __restrict__ B,
                         float* __restrict__ C, int Ncols, int K) {
  const int wave = threadIdx.x >> 5;
  const int lane = threadIdx.x & 31;
  const int row0 = blockIdx.x * 128 + wave * 16;
  const int col0 = blockIdx.y * 64;
  const int half = lane >> 4, n = lane & 15;

  const __bf16* pfA  = A + (size_t)(row0 + n) * K;         // rows row0..row0+15
  const __bf16* pfB0 = B + (size_t)(col0 + lane) * K;      // rows col0..col0+31
  const __bf16* pfB1 = B + (size_t)(col0 + 32 + lane) * K; // rows col0+32..col0+63

  v8f c0 = {}, c1 = {}, c2 = {}, c3 = {};
  for (int kk = 0; kk < K; kk += 64) {
    // WGP-scope prefetch of the K tile two iterations ahead (speculative
    // prefetches past the tensor end are silently dropped per ISA)
    prefetch_wgp(pfA + kk + 128);
    prefetch_wgp(pfB0 + kk + 128);
    prefetch_wgp(pfB1 + kk + 128);
    gemm_step(A, B, K, row0, col0, kk,      lane, c0, c1, c2, c3);
    gemm_step(A, B, K, row0, col0, kk + 32, lane, c0, c1, c2, c3);
  }

#pragma unroll
  for (int r = 0; r < 8; ++r) {
    float* crow = C + (size_t)(row0 + r + 8 * half) * Ncols + n;
    crow[col0]      = c0[r];
    crow[col0 + 16] = c1[r];
    crow[col0 + 32] = c2[r];
    crow[col0 + 48] = c3[r];
  }
}

// ---------------------------------------------------------------------------
// BN stats, two-phase for parallelism: partial sums over 16 row chunks with
// fp32 atomics, then finalize mu / rsqrt(var+eps).
// ---------------------------------------------------------------------------
__global__ void bn_partial(const float* __restrict__ h, float* __restrict__ sum,
                           float* __restrict__ sumsq, int H, int rowsPer) {
  int col = blockIdx.x * blockDim.x + threadIdx.x;
  int r0 = blockIdx.y * rowsPer;
  float s = 0.f, s2 = 0.f;
  for (int r = r0; r < r0 + rowsPer; ++r) {
    float v = h[(size_t)r * H + col];
    s += v; s2 += v * v;
  }
  atomicAdd(&sum[col], s);
  atomicAdd(&sumsq[col], s2);
}

__global__ void bn_finalize(const float* __restrict__ sum, const float* __restrict__ sumsq,
                            float* __restrict__ mu, float* __restrict__ invstd, int N, int H) {
  int col = blockIdx.x * blockDim.x + threadIdx.x;
  if (col >= H) return;
  float m = sum[col] / (float)N;
  mu[col] = m;
  invstd[col] = rsqrtf(sumsq[col] / (float)N - m * m + BN_EPS_C);
}

// ---------------------------------------------------------------------------
// BN apply + ReLU, fused with bf16 conversion feeding GEMM2.
// ---------------------------------------------------------------------------
__global__ void bn_apply_relu(const float* __restrict__ h, const float* __restrict__ mu,
                              const float* __restrict__ invstd, const float* __restrict__ g,
                              const float* __restrict__ b, __bf16* __restrict__ out,
                              int H, size_t total) {
  size_t i = (size_t)blockIdx.x * blockDim.x + threadIdx.x;
  if (i >= total) return;
  int col = (int)(i % H);
  float v = (h[i] - mu[col]) * invstd[col] * g[col] + b[col];
  out[i] = (__bf16)fmaxf(v, 0.0f);
}

// ---------------------------------------------------------------------------
// Row L2 normalize -> bf16. One block (128 threads) per row, P == 128.
// ---------------------------------------------------------------------------
__global__ void l2norm_bf16(const float* __restrict__ z, __bf16* __restrict__ zn, int P) {
  const int row = blockIdx.x, t = threadIdx.x;
  float v = z[(size_t)row * P + t];
  __shared__ float red[128];
  red[t] = v * v;
  __syncthreads();
  for (int s = 64; s > 0; s >>= 1) {
    if (t < s) red[t] += red[t + s];
    __syncthreads();
  }
  float invn = rsqrtf(red[0]);
  zn[(size_t)row * P + t] = (__bf16)(v * invn);
}

// ---------------------------------------------------------------------------
// Fused similarity + RBF weights + streaming log-softmax loss.
// Never materializes the 2Nx2N sim or weight matrices.
// Per row r: loss_r = log(sum_c exp(sim_rc)) - (sum_c w~ * sim_rc)/(sum_c w~)
// (no max-subtraction needed: sim in [-10,10] except diag = -1e8 -> exp = 0)
// A fragments (the row block) are invariant across column tiles -> hoisted.
// grid = 2N/16 row tiles; block = 256 (8 waves striping 512 column tiles).
// ---------------------------------------------------------------------------
__global__ void sim_loss_kernel(const __bf16* __restrict__ zn, const float* __restrict__ labels,
                                float* __restrict__ out, int twoN, int Nref, int P) {
  const int wave = threadIdx.x >> 5, lane = threadIdx.x & 31;
  const int half = lane >> 4, n = lane & 15;
  const int row0 = blockIdx.x * 16;
  const int nct = twoN >> 4;

  v16bf afr[4];
#pragma unroll
  for (int i = 0; i < 4; ++i) afr[i] = load_frag_a(zn, P, row0, 32 * i, lane);

  float accS[8], accT[8], accW[8], labr[8];
#pragma unroll
  for (int e = 0; e < 8; ++e) {
    accS[e] = 0.f; accT[e] = 0.f; accW[e] = 0.f;
    labr[e] = labels[(row0 + e + 8 * half) % Nref];
  }

  for (int ct = wave; ct < nct; ct += 8) {          // wave-uniform trip count
    const int col0 = ct * 16;
    v8f c = {};
#pragma unroll
    for (int i = 0; i < 4; ++i) {
      v16bf b = load_frag_b(zn, P, col0, 32 * i, lane);
      c = WMMA_BF16(afr[i], b, c);
    }
    const float labc = labels[(col0 + n) % Nref];
#pragma unroll
    for (int e = 0; e < 8; ++e) {
      const int gr = row0 + e + 8 * half;
      const int gc = col0 + n;
      const bool diag = (gr == gc);
      float sim = c[e] * TEMP_INV;
      if (diag) sim -= INF_C;
      const float d = labr[e] - labc;
      const float w = diag ? 0.f : expf(-GAMMA_C * d * d);
      accS[e] += expf(sim);
      accT[e] += w * sim;
      accW[e] += w;
    }
  }

  // reduce over columns: 16 lanes within each half (xor masks 1..8 stay in-half)
#pragma unroll
  for (int e = 0; e < 8; ++e) {
#pragma unroll
    for (int m = 8; m >= 1; m >>= 1) {
      accS[e] += __shfl_xor(accS[e], m, 32);
      accT[e] += __shfl_xor(accT[e], m, 32);
      accW[e] += __shfl_xor(accW[e], m, 32);
    }
  }

  __shared__ float sS[8][16], sT[8][16], sW[8][16];
  if (n == 0) {
#pragma unroll
    for (int e = 0; e < 8; ++e) {
      sS[wave][e + 8 * half] = accS[e];
      sT[wave][e + 8 * half] = accT[e];
      sW[wave][e + 8 * half] = accW[e];
    }
  }
  __syncthreads();
  if (threadIdx.x < 16) {
    float S = 0.f, T = 0.f, W = 0.f;
#pragma unroll
    for (int w = 0; w < 8; ++w) { S += sS[w][threadIdx.x]; T += sT[w][threadIdx.x]; W += sW[w][threadIdx.x]; }
    const float contrib = logf(S) - T / W;
    atomicAdd(out, contrib / (float)Nref);
  }
}

// ---------------------------------------------------------------------------
extern "C" void kernel_launch(void* const* d_in, const int* in_sizes, int n_in,
                              void* d_out, int out_size, void* d_ws, size_t ws_size,
                              hipStream_t stream) {
  (void)n_in; (void)out_size; (void)ws_size;
  const float* y1  = (const float*)d_in[0];
  const float* y2  = (const float*)d_in[1];
  const float* lab = (const float*)d_in[2];
  const float* W1  = (const float*)d_in[3];
  const float* bng = (const float*)d_in[4];
  const float* bnb = (const float*)d_in[5];
  const float* W2  = (const float*)d_in[6];

  const int N = in_sizes[2];            // 4096
  const int D = in_sizes[0] / N;        // 2048
  const int H = in_sizes[4];            // 2048
  const int P = in_sizes[6] / H;        // 128
  const int twoN = 2 * N;

  char* w = (char*)d_ws;
  auto carve = [&](size_t bytes) -> char* {
    char* p = w; w += (bytes + 255) & ~(size_t)255; return p;
  };
  __bf16* y1b = (__bf16*)carve((size_t)N * D * 2);
  __bf16* y2b = (__bf16*)carve((size_t)N * D * 2);
  __bf16* W1b = (__bf16*)carve((size_t)H * D * 2);
  __bf16* W2b = (__bf16*)carve((size_t)P * H * 2);
  float*  h1  = (float*) carve((size_t)N * H * 4);
  float*  h2  = (float*) carve((size_t)N * H * 4);
  __bf16* hb1 = (__bf16*)carve((size_t)N * H * 2);
  __bf16* hb2 = (__bf16*)carve((size_t)N * H * 2);
  float*  st  = (float*) carve((size_t)4 * H * 4);  // [s1 | q1 | s2 | q2]
  float*  mu1 = (float*) carve((size_t)H * 4);
  float*  iv1 = (float*) carve((size_t)H * 4);
  float*  mu2 = (float*) carve((size_t)H * 4);
  float*  iv2 = (float*) carve((size_t)H * 4);
  float*  zc  = (float*) carve((size_t)twoN * P * 4);
  __bf16* znb = (__bf16*)carve((size_t)twoN * P * 2);
  float* s1 = st, *q1 = st + H, *s2 = st + 2 * H, *q2 = st + 3 * H;

  zero_f32<<<1, 32, 0, stream>>>((float*)d_out, 1);
  zero_f32<<<(4 * H + 255) / 256, 256, 0, stream>>>(st, 4 * H);

  const size_t nYD = (size_t)N * D, nHD = (size_t)H * D, nPH = (size_t)P * H;
  cvt_bf16<<<(unsigned)((nYD + 255) / 256), 256, 0, stream>>>(y1, y1b, nYD);
  cvt_bf16<<<(unsigned)((nYD + 255) / 256), 256, 0, stream>>>(y2, y2b, nYD);
  cvt_bf16<<<(unsigned)((nHD + 255) / 256), 256, 0, stream>>>(W1, W1b, nHD);
  cvt_bf16<<<(unsigned)((nPH + 255) / 256), 256, 0, stream>>>(W2, W2b, nPH);

  dim3 blk(256);
  dim3 g1(N / 128, H / 64);
  gemm_xwT<<<g1, blk, 0, stream>>>(y1b, W1b, h1, H, D);
  gemm_xwT<<<g1, blk, 0, stream>>>(y2b, W1b, h2, H, D);

  const int RCHUNK = 16;
  dim3 gs(H / 256, RCHUNK);
  bn_partial<<<gs, 256, 0, stream>>>(h1, s1, q1, H, N / RCHUNK);
  bn_partial<<<gs, 256, 0, stream>>>(h2, s2, q2, H, N / RCHUNK);
  bn_finalize<<<(H + 255) / 256, 256, 0, stream>>>(s1, q1, mu1, iv1, N, H);
  bn_finalize<<<(H + 255) / 256, 256, 0, stream>>>(s2, q2, mu2, iv2, N, H);

  const size_t nNH = (size_t)N * H;
  bn_apply_relu<<<(unsigned)((nNH + 255) / 256), 256, 0, stream>>>(h1, mu1, iv1, bng, bnb, hb1, H, nNH);
  bn_apply_relu<<<(unsigned)((nNH + 255) / 256), 256, 0, stream>>>(h2, mu2, iv2, bng, bnb, hb2, H, nNH);

  dim3 g2(N / 128, P / 64);
  gemm_xwT<<<g2, blk, 0, stream>>>(hb1, W2b, zc, P, H);
  gemm_xwT<<<g2, blk, 0, stream>>>(hb2, W2b, zc + (size_t)N * P, P, H);

  l2norm_bf16<<<twoN, 128, 0, stream>>>(zc, znb, P);

  sim_loss_kernel<<<twoN / 16, 256, 0, stream>>>(znb, lab, (float*)d_out, twoN, N, P);
}
